// S2GraphBlock_14181982011505
// MI455X (gfx1250) — matrix operand order repaired and checked by reading, not compiled, not run on hardware
//
#include <hip/hip_runtime.h>

#define B_    8
#define N_    1024
#define D_    256
#define E_    32768
#define EF_   16
#define ROWS_ (B_*N_)
#define NEGMIN (-3.402823466e38f)

typedef __attribute__((ext_vector_type(16))) __bf16 v16bf;
typedef __attribute__((ext_vector_type(8)))  float  v8f;

union FragBF { v16bf v; uint4 q[2]; unsigned short u[16]; };
union F8     { v8f v; float f[8]; };
union U4     { uint4 q; unsigned short u[8]; };

__device__ __forceinline__ unsigned short f2bf(float f) {
  unsigned u = __float_as_uint(f);
  u += 0x7FFFu + ((u >> 16) & 1u);           // round-to-nearest-even
  return (unsigned short)(u >> 16);
}

__device__ __forceinline__ v8f wmma_bf(const FragBF& a, const FragBF& b, v8f c) {
  return __builtin_amdgcn_wmma_f32_16x16x32_bf16(false, a.v, false, b.v,
                                                 (short)0, c, false, false);
}

// reductions across the 16 lanes holding one output row (masks stay in half-wave)
__device__ __forceinline__ float red_sum16(float x) {
  x += __shfl_xor(x, 1); x += __shfl_xor(x, 2);
  x += __shfl_xor(x, 4); x += __shfl_xor(x, 8);
  return x;
}
__device__ __forceinline__ float red_max16(float x) {
  x = fmaxf(x, __shfl_xor(x, 1)); x = fmaxf(x, __shfl_xor(x, 2));
  x = fmaxf(x, __shfl_xor(x, 4)); x = fmaxf(x, __shfl_xor(x, 8));
  return x;
}

struct GeoI { float x1,y1,x2,y2,cx,cy,lw,lh,area,inva; };
__device__ __forceinline__ GeoI load_geo(const float* g) {
  GeoI r; r.x1=g[0]; r.y1=g[1]; r.x2=g[2]; r.y2=g[3]; r.cx=g[4]; r.cy=g[5];
  r.lw=g[6]; r.lh=g[7]; r.area=g[8]; r.inva=g[9]; return r;
}
__device__ __forceinline__ float geo_bias(const GeoI& a, const GeoI& b,
                                          const float* gw, float gb) {
  float dx = a.cx - b.cx, dy = a.cy - b.cy;
  float f2 = a.lw - b.lw, f3 = a.lh - b.lh;
  float dist = sqrtf(dx*dx + dy*dy + 1e-6f);
  float ix1 = fmaxf(a.x1, b.x1), iy1 = fmaxf(a.y1, b.y1);
  float ix2 = fminf(a.x2, b.x2), iy2 = fminf(a.y2, b.y2);
  float iw = fmaxf(ix2-ix1, 0.f), ih = fmaxf(iy2-iy1, 0.f);
  float inter = iw*ih;
  float iou = inter / (a.area + b.area - inter + 1e-6f);
  float ratio = a.area * b.inva;
  return gw[0]*dx + gw[1]*dy + gw[2]*f2 + gw[3]*f3 +
         gw[4]*dist + gw[5]*iou + gw[6]*ratio + gb;
}

// ---------------------------------------------------------------- pack kernels
__global__ void cvt_bf16_kernel(const float* __restrict__ s,
                                unsigned short* __restrict__ d, int n) {
  int i = blockIdx.x*blockDim.x + threadIdx.x;
  if (i < n) d[i] = f2bf(s[i]);
}

// dst[n*K + k] = src[k*Nn + n] * scale  (row-major K x Nn -> transposed bf16)
__global__ void packT_kernel(const float* __restrict__ src,
                             unsigned short* __restrict__ dst,
                             int K, int Nn, float scale) {
  int i = blockIdx.x*blockDim.x + threadIdx.x;
  if (i >= K*Nn) return;
  int n = i / K, k = i - n*K;
  dst[i] = f2bf(src[k*Nn + n] * scale);
}

// fused Wq|Wk|Wv transpose-pack; 1/sqrt(D)=1/16 folded into Wq & bq
__global__ void pack_qkvT_kernel(const float* __restrict__ Wq, const float* __restrict__ Wk,
                                 const float* __restrict__ Wv, const float* __restrict__ bq,
                                 const float* __restrict__ bk, const float* __restrict__ bv,
                                 unsigned short* __restrict__ WT, float* __restrict__ bqkv) {
  int i = blockIdx.x*blockDim.x + threadIdx.x;
  if (i >= 768*256) return;
  int n = i >> 8, k = i & 255;
  const float* W; float sc; int c;
  if (n < 256)      { W = Wq; sc = 0.0625f; c = n; }
  else if (n < 512) { W = Wk; sc = 1.f;     c = n-256; }
  else              { W = Wv; sc = 1.f;     c = n-512; }
  WT[i] = f2bf(W[k*256 + c] * sc);
  if (k == 0)
    bqkv[n] = (n < 256) ? bq[n]*0.0625f : (n < 512 ? bk[n-256] : bv[n-512]);
}

// per-node geometry features (log/area precomputed so per-pair bias is cheap)
__global__ void geo_kernel(const float* __restrict__ boxes, float* __restrict__ ng) {
  int i = blockIdx.x*blockDim.x + threadIdx.x;
  if (i >= ROWS_) return;
  float x1 = boxes[i*4+0], y1 = boxes[i*4+1], x2 = boxes[i*4+2], y2 = boxes[i*4+3];
  float w = fmaxf(x2-x1, 1e-6f), h = fmaxf(y2-y1, 1e-6f);
  float area = w*h;
  float* g = ng + (size_t)i*12;
  g[0]=x1; g[1]=y1; g[2]=x2; g[3]=y2;
  g[4]=(x1+x2)*0.5f; g[5]=(y1+y2)*0.5f;
  g[6]=logf(w); g[7]=logf(h);
  g[8]=area; g[9]=1.f/fmaxf(area,1e-6f); g[10]=0.f; g[11]=0.f;
}

__global__ void zero_kernel(uint4* __restrict__ p, int n4) {
  for (int i = blockIdx.x*blockDim.x + threadIdx.x; i < n4; i += gridDim.x*blockDim.x)
    p[i] = make_uint4(0u,0u,0u,0u);
}

// sparse edge bias + adjacency bitmask scatter
__global__ void scatter_kernel(const int* __restrict__ ei, const float* __restrict__ ea,
                               const float* __restrict__ ew, const float* __restrict__ eb,
                               float* __restrict__ sp, unsigned* __restrict__ adj) {
  int idx = blockIdx.x*blockDim.x + threadIdx.x;
  if (idx >= B_*E_) return;
  int b = idx >> 15, e = idx & (E_-1);
  int s = ei[(b*2)*E_ + e], d = ei[(b*2+1)*E_ + e];
  if (s < 0 || d < 0 || s >= N_ || d >= N_) return;
  float p = eb[0];
  const float* a = ea + (size_t)(b*E_ + e)*EF_;
  #pragma unroll
  for (int f = 0; f < EF_; ++f) p += a[f]*ew[f];
  sp[((size_t)(b*N_ + s))*N_ + d] = p;
  atomicOr(&adj[(size_t)(b*N_ + s)*32 + (d >> 5)], 1u << (d & 31));
}

// ---------------------------------------------------------------- QKV GEMM
// block = 4 waves; each wave: 16 rows x 64 cols of [8192 x 768]
__global__ __launch_bounds__(128) void qkv_kernel(
    const unsigned short* __restrict__ nbf, const unsigned short* __restrict__ WT,
    const float* __restrict__ bqkv, unsigned short* __restrict__ qbf,
    unsigned short* __restrict__ kbf, unsigned short* __restrict__ vbf) {
  const int lane = threadIdx.x & 31, wv = threadIdx.x >> 5;
  const int col = lane & 15, hi = lane >> 4;
  const int rb = blockIdx.x*64 + wv*16;
  const int nb = blockIdx.y*64;
  F8 acc[4];
  #pragma unroll
  for (int t = 0; t < 4; ++t) {
    #pragma unroll
    for (int e = 0; e < 8; ++e) acc[t].f[e] = 0.f;
  }
  const size_t arow = (size_t)(rb + col)*D_;
  #pragma unroll
  for (int kk = 0; kk < 8; ++kk) {
    FragBF af;
    const unsigned short* ap = nbf + arow + kk*32 + hi*16;
    af.q[0] = *(const uint4*)ap; af.q[1] = *(const uint4*)(ap + 8);
    #pragma unroll
    for (int t = 0; t < 4; ++t) {
      FragBF bf;
      const unsigned short* bp = WT + (size_t)(nb + t*16 + col)*D_ + kk*32 + hi*16;
      bf.q[0] = *(const uint4*)bp; bf.q[1] = *(const uint4*)(bp + 8);
      acc[t].v = wmma_bf(af, bf, acc[t].v);
    }
  }
  #pragma unroll
  for (int t = 0; t < 4; ++t) {
    const int n = nb + t*16 + col;
    const float bia = bqkv[n];
    #pragma unroll
    for (int v = 0; v < 8; ++v) {
      const int r = rb + hi*8 + v;
      unsigned short hbit = f2bf(acc[t].f[v] + bia);
      if (n < 256)      qbf[(size_t)r*D_ + n]       = hbit;
      else if (n < 512) kbf[(size_t)r*D_ + (n-256)] = hbit;
      else              vbf[(size_t)r*D_ + (n-512)] = hbit;
    }
  }
}

// ---------------------------------------------------------------- flash attention
// grid (N/64, B), 4 waves x 16 rows; streams 32-col K/V tiles through LDS
__global__ __launch_bounds__(128) void attn_kernel(
    const unsigned short* __restrict__ qbf, const unsigned short* __restrict__ kbf,
    const unsigned short* __restrict__ vbf, const float* __restrict__ ngeo,
    const float* __restrict__ spb, const unsigned* __restrict__ adj,
    const unsigned char* __restrict__ nmask, const float* __restrict__ geo_w,
    const float* __restrict__ geo_b, unsigned short* __restrict__ obf) {
  __shared__ __align__(16) unsigned short kt[32*256];   // K tile row-major [j][k]
  __shared__ __align__(16) unsigned short vt[256*32];   // V tile transposed [d][j]
  __shared__ __align__(16) unsigned short pbw[4][16*32];

  const int lane = threadIdx.x & 31, wv = threadIdx.x >> 5;
  const int col = lane & 15, hi = lane >> 4;
  const int b = blockIdx.y;
  const int rb = blockIdx.x*64 + wv*16;

  float gw[7];
  #pragma unroll
  for (int f = 0; f < 7; ++f) gw[f] = geo_w[f];
  const float gb = geo_b[0];

  FragBF qf[8];
  {
    const size_t qrow = (size_t)(b*N_ + rb + col)*D_;
    #pragma unroll
    for (int kk = 0; kk < 8; ++kk) {
      const unsigned short* p = qbf + qrow + kk*32 + hi*16;
      qf[kk].q[0] = *(const uint4*)p; qf[kk].q[1] = *(const uint4*)(p + 8);
    }
  }
  GeoI ig[8]; bool mi[8];
  #pragma unroll
  for (int v = 0; v < 8; ++v) {
    const int i = rb + hi*8 + v;
    ig[v] = load_geo(ngeo + (size_t)(b*N_ + i)*12);
    mi[v] = nmask[b*N_ + i] != 0;
  }
  float m[8], l[8];
  #pragma unroll
  for (int v = 0; v < 8; ++v) { m[v] = NEGMIN; l[v] = 0.f; }
  F8 acc[16];
  #pragma unroll
  for (int t = 0; t < 16; ++t) {
    #pragma unroll
    for (int e = 0; e < 8; ++e) acc[t].f[e] = 0.f;
  }

  for (int jb = 0; jb < N_; jb += 32) {
    __syncthreads();
    { // stage K (row-major) and V (transposed) tiles
      const uint4* ks = (const uint4*)(kbf + (size_t)(b*N_ + jb)*D_);
      uint4* kd = (uint4*)kt;
      #pragma unroll
      for (int t = 0; t < 8; ++t) kd[threadIdx.x + 128*t] = ks[threadIdx.x + 128*t];
      const unsigned short* vs = vbf + (size_t)(b*N_ + jb)*D_;
      #pragma unroll
      for (int t = 0; t < 8; ++t) {
        const int idx = threadIdx.x + 128*t;
        const int r = idx >> 5, cb = (idx & 31)*8;
        U4 d4; d4.q = *(const uint4*)(vs + (size_t)r*D_ + cb);
        #pragma unroll
        for (int e = 0; e < 8; ++e) vt[(cb + e)*32 + r] = d4.u[e];
      }
      if (jb + 32 < N_) {                       // pull next tiles toward L2/L0
        __builtin_prefetch(kbf + (size_t)(b*N_ + jb + 32)*D_ + (size_t)threadIdx.x*128, 0, 1);
        __builtin_prefetch(vbf + (size_t)(b*N_ + jb + 32)*D_ + (size_t)threadIdx.x*128, 0, 1);
      }
    }
    __syncthreads();

    F8 s0, s1;
    #pragma unroll
    for (int e = 0; e < 8; ++e) { s0.f[e] = 0.f; s1.f[e] = 0.f; }
    #pragma unroll
    for (int kk = 0; kk < 8; ++kk) {            // S = Q * K^T  (two 16x16 tiles)
      FragBF b0, b1;
      const unsigned short* p0 = &kt[col*256 + kk*32 + hi*16];
      b0.q[0] = *(const uint4*)p0; b0.q[1] = *(const uint4*)(p0 + 8);
      const unsigned short* p1 = &kt[(16 + col)*256 + kk*32 + hi*16];
      b1.q[0] = *(const uint4*)p1; b1.q[1] = *(const uint4*)(p1 + 8);
      s0.v = wmma_bf(qf[kk], b0, s0.v);
      s1.v = wmma_bf(qf[kk], b1, s1.v);
    }

    const int j0 = jb + col, j1 = jb + 16 + col;
    const GeoI jg0 = load_geo(ngeo + (size_t)(b*N_ + j0)*12);
    const GeoI jg1 = load_geo(ngeo + (size_t)(b*N_ + j1)*12);
    const bool mj0 = nmask[b*N_ + j0] != 0, mj1 = nmask[b*N_ + j1] != 0;

    float p0[8], p1[8], al[8];
    #pragma unroll
    for (int v = 0; v < 8; ++v) {               // online softmax (per row)
      const int i = rb + hi*8 + v;
      const size_t ib = (size_t)(b*N_ + i);
      const unsigned w = adj[ib*32 + (jb >> 5)];
      const bool a0 = mi[v] && mj0 && (((w >> col) & 1u) || (i == j0));
      const bool a1 = mi[v] && mj1 && (((w >> (16 + col)) & 1u) || (i == j1));
      const float bias0 = geo_bias(ig[v], jg0, gw, gb) + spb[ib*N_ + j0];
      const float bias1 = geo_bias(ig[v], jg1, gw, gb) + spb[ib*N_ + j1];
      const float sv0 = a0 ? s0.f[v] + bias0 : NEGMIN;
      const float sv1 = a1 ? s1.f[v] + bias1 : NEGMIN;
      const float mx = red_max16(fmaxf(sv0, sv1));
      const float mn = fmaxf(m[v], mx);
      const float a_ = __expf(m[v] - mn);       // 1.0 when nothing valid yet
      const float e0 = a0 ? __expf(sv0 - mn) : 0.f;
      const float e1 = a1 ? __expf(sv1 - mn) : 0.f;
      l[v] = l[v]*a_ + red_sum16(e0 + e1);
      m[v] = mn; al[v] = a_; p0[v] = e0; p1[v] = e1;
    }
    #pragma unroll
    for (int t = 0; t < 16; ++t) {
      #pragma unroll
      for (int v = 0; v < 8; ++v) acc[t].f[v] *= al[v];
    }

    // C/D-layout P -> LDS -> A-fragment layout (wave-private, DS is in-order)
    #pragma unroll
    for (int v = 0; v < 8; ++v) {
      const int pr = hi*8 + v;
      pbw[wv][pr*32 + col]      = f2bf(p0[v]);
      pbw[wv][pr*32 + 16 + col] = f2bf(p1[v]);
    }
    FragBF pf;
    const unsigned short* pp = &pbw[wv][col*32 + hi*16];
    pf.q[0] = *(const uint4*)pp; pf.q[1] = *(const uint4*)(pp + 8);

    #pragma unroll
    for (int dt = 0; dt < 16; ++dt) {           // O += P * V
      FragBF vf;
      const unsigned short* vp = &vt[(dt*16 + col)*32 + hi*16];
      vf.q[0] = *(const uint4*)vp; vf.q[1] = *(const uint4*)(vp + 8);
      acc[dt].v = wmma_bf(pf, vf, acc[dt].v);
    }
  }

  float inv[8];
  #pragma unroll
  for (int v = 0; v < 8; ++v) inv[v] = (l[v] > 0.f) ? 1.f/l[v] : 0.f;
  #pragma unroll
  for (int dt = 0; dt < 16; ++dt) {
    #pragma unroll
    for (int v = 0; v < 8; ++v)
      obf[(size_t)(b*N_ + rb + hi*8 + v)*D_ + dt*16 + col] = f2bf(acc[dt].f[v]*inv[v]);
  }
}

// ---------------------------------------------------------------- out-proj + LN1
__global__ __launch_bounds__(128) void proj_ln1_kernel(
    const unsigned short* __restrict__ abf, const unsigned short* __restrict__ WoT,
    const float* __restrict__ bo, const float* __restrict__ nodes,
    const float* __restrict__ g1, const float* __restrict__ b1,
    float* __restrict__ xf32, unsigned short* __restrict__ xbf) {
  const int lane = threadIdx.x & 31, wv = threadIdx.x >> 5;
  const int col = lane & 15, hi = lane >> 4;
  const int rb = blockIdx.x*64 + wv*16;
  F8 acc[16];
  #pragma unroll
  for (int t = 0; t < 16; ++t) {
    #pragma unroll
    for (int e = 0; e < 8; ++e) acc[t].f[e] = 0.f;
  }
  const size_t arow = (size_t)(rb + col)*D_;
  #pragma unroll
  for (int kk = 0; kk < 8; ++kk) {
    FragBF af;
    const unsigned short* ap = abf + arow + kk*32 + hi*16;
    af.q[0] = *(const uint4*)ap; af.q[1] = *(const uint4*)(ap + 8);
    #pragma unroll
    for (int t = 0; t < 16; ++t) {
      FragBF bf;
      const unsigned short* bp = WoT + (size_t)(t*16 + col)*D_ + kk*32 + hi*16;
      bf.q[0] = *(const uint4*)bp; bf.q[1] = *(const uint4*)(bp + 8);
      acc[t].v = wmma_bf(af, bf, acc[t].v);
    }
  }
  #pragma unroll
  for (int v = 0; v < 8; ++v) {                 // residual + LayerNorm
    const int row = rb + hi*8 + v;
    float tmp[16], s = 0.f;
    #pragma unroll
    for (int t = 0; t < 16; ++t) {
      const int c = t*16 + col;
      const float x = acc[t].f[v] + bo[c] + nodes[(size_t)row*D_ + c];
      tmp[t] = x; s += x;
    }
    const float mean = red_sum16(s) * (1.f/256.f);
    float s2 = 0.f;
    #pragma unroll
    for (int t = 0; t < 16; ++t) { const float d = tmp[t]-mean; s2 += d*d; }
    const float rstd = rsqrtf(red_sum16(s2)*(1.f/256.f) + 1e-5f);
    #pragma unroll
    for (int t = 0; t < 16; ++t) {
      const int c = t*16 + col;
      const float y = (tmp[t]-mean)*rstd*g1[c] + b1[c];
      xf32[(size_t)row*D_ + c] = y;
      xbf[(size_t)row*D_ + c]  = f2bf(y);
    }
  }
}

// ---------------------------------------------------------------- fused FFN + LN2
__global__ __launch_bounds__(128) void ffn_kernel(
    const unsigned short* __restrict__ xbf, const float* __restrict__ xf32,
    const unsigned short* __restrict__ W1T, const float* __restrict__ b1f,
    const unsigned short* __restrict__ W2T, const float* __restrict__ b2f,
    const float* __restrict__ g2, const float* __restrict__ be2,
    const unsigned char* __restrict__ nmask, float* __restrict__ out) {
  __shared__ __align__(16) unsigned short pbuf[4][16*32];
  const int lane = threadIdx.x & 31, wv = threadIdx.x >> 5;
  const int col = lane & 15, hi = lane >> 4;
  const int rb = blockIdx.x*64 + wv*16;

  FragBF xf[8];
  {
    const size_t arow = (size_t)(rb + col)*D_;
    #pragma unroll
    for (int kk = 0; kk < 8; ++kk) {
      const unsigned short* p = xbf + arow + kk*32 + hi*16;
      xf[kk].q[0] = *(const uint4*)p; xf[kk].q[1] = *(const uint4*)(p + 8);
    }
  }
  F8 y[16];
  #pragma unroll
  for (int t = 0; t < 16; ++t) {
    #pragma unroll
    for (int e = 0; e < 8; ++e) y[t].f[e] = 0.f;
  }

  for (int c = 0; c < 32; ++c) {                // 32 hidden-cols per chunk
    F8 h0, h1;
    #pragma unroll
    for (int e = 0; e < 8; ++e) { h0.f[e] = 0.f; h1.f[e] = 0.f; }
    const int hc0 = c*32 + col, hc1 = hc0 + 16;
    #pragma unroll
    for (int kk = 0; kk < 8; ++kk) {
      FragBF w0, w1;
      const unsigned short* p0 = W1T + (size_t)hc0*D_ + kk*32 + hi*16;
      w0.q[0] = *(const uint4*)p0; w0.q[1] = *(const uint4*)(p0 + 8);
      const unsigned short* p1 = W1T + (size_t)hc1*D_ + kk*32 + hi*16;
      w1.q[0] = *(const uint4*)p1; w1.q[1] = *(const uint4*)(p1 + 8);
      h0.v = wmma_bf(xf[kk], w0, h0.v);
      h1.v = wmma_bf(xf[kk], w1, h1.v);
    }
    const float bb0 = b1f[hc0], bb1 = b1f[hc1];
    #pragma unroll
    for (int v = 0; v < 8; ++v) {               // exact GELU, stage to LDS
      const int pr = hi*8 + v;
      const float a0 = h0.f[v] + bb0, a1 = h1.f[v] + bb1;
      pbuf[wv][pr*32 + col]      = f2bf(0.5f*a0*(1.f + erff(a0*0.70710678118f)));
      pbuf[wv][pr*32 + 16 + col] = f2bf(0.5f*a1*(1.f + erff(a1*0.70710678118f)));
    }
    FragBF hf;
    const unsigned short* pp = &pbuf[wv][col*32 + hi*16];
    hf.q[0] = *(const uint4*)pp; hf.q[1] = *(const uint4*)(pp + 8);
    #pragma unroll
    for (int dt = 0; dt < 16; ++dt) {
      FragBF w2;
      const unsigned short* wp = W2T + (size_t)(dt*16 + col)*1024 + c*32 + hi*16;
      w2.q[0] = *(const uint4*)wp; w2.q[1] = *(const uint4*)(wp + 8);
      y[dt].v = wmma_bf(hf, w2, y[dt].v);
    }
  }
  #pragma unroll
  for (int v = 0; v < 8; ++v) {                 // residual + LN2 + mask
    const int row = rb + hi*8 + v;
    float tmp[16], s = 0.f;
    #pragma unroll
    for (int t = 0; t < 16; ++t) {
      const int cc = t*16 + col;
      const float x = y[t].f[v] + b2f[cc] + xf32[(size_t)row*D_ + cc];
      tmp[t] = x; s += x;
    }
    const float mean = red_sum16(s) * (1.f/256.f);
    float s2 = 0.f;
    #pragma unroll
    for (int t = 0; t < 16; ++t) { const float d = tmp[t]-mean; s2 += d*d; }
    const float rstd = rsqrtf(red_sum16(s2)*(1.f/256.f) + 1e-5f);
    const float msk = nmask[row] ? 1.f : 0.f;
    #pragma unroll
    for (int t = 0; t < 16; ++t) {
      const int cc = t*16 + col;
      out[(size_t)row*D_ + cc] = msk*((tmp[t]-mean)*rstd*g2[cc] + be2[cc]);
    }
  }
}

// ---------------------------------------------------------------- launch
extern "C" void kernel_launch(void* const* d_in, const int* in_sizes, int n_in,
                              void* d_out, int out_size, void* d_ws, size_t ws_size,
                              hipStream_t stream) {
  (void)in_sizes; (void)n_in; (void)out_size; (void)ws_size;
  const float* nodes  = (const float*)d_in[0];
  const float* boxes  = (const float*)d_in[1];
  const float* eattr  = (const float*)d_in[2];
  const float* Wq = (const float*)d_in[3];  const float* bq = (const float*)d_in[4];
  const float* Wk = (const float*)d_in[5];  const float* bk = (const float*)d_in[6];
  const float* Wv = (const float*)d_in[7];  const float* bv = (const float*)d_in[8];
  const float* Wo = (const float*)d_in[9];  const float* bo = (const float*)d_in[10];
  const float* geo_w = (const float*)d_in[11]; const float* geo_b = (const float*)d_in[12];
  const float* edge_w = (const float*)d_in[13]; const float* edge_b = (const float*)d_in[14];
  const float* ln1g = (const float*)d_in[15]; const float* ln1b = (const float*)d_in[16];
  const float* ln2g = (const float*)d_in[17]; const float* ln2b = (const float*)d_in[18];
  const float* W1 = (const float*)d_in[19]; const float* b1f = (const float*)d_in[20];
  const float* W2 = (const float*)d_in[21]; const float* b2f = (const float*)d_in[22];
  const unsigned char* nmask = (const unsigned char*)d_in[23];
  const int* eidx = (const int*)d_in[24];
  float* out = (float*)d_out;

  char* w = (char*)d_ws;
  constexpr size_t SZ_RD_BF = (size_t)ROWS_*D_*2;          // 4 MiB
  size_t o = 0;
  unsigned short* nbf  = (unsigned short*)(w + o); o += SZ_RD_BF;
  unsigned short* qbf  = (unsigned short*)(w + o); o += SZ_RD_BF;
  unsigned short* kbf  = (unsigned short*)(w + o); o += SZ_RD_BF;
  unsigned short* vbf  = (unsigned short*)(w + o); o += SZ_RD_BF;
  unsigned short* abf  = (unsigned short*)(w + o); o += SZ_RD_BF;
  unsigned short* xbf  = (unsigned short*)(w + o); o += SZ_RD_BF;
  float*          xf32 = (float*)(w + o);          o += (size_t)ROWS_*D_*4;
  unsigned short* WqkvT= (unsigned short*)(w + o); o += (size_t)768*256*2;
  float*          bqkv = (float*)(w + o);          o += 768*4;
  unsigned short* WoT  = (unsigned short*)(w + o); o += (size_t)256*256*2;
  unsigned short* W1T  = (unsigned short*)(w + o); o += (size_t)1024*256*2;
  unsigned short* W2T  = (unsigned short*)(w + o); o += (size_t)256*1024*2;
  float*          ngeo = (float*)(w + o);          o += (size_t)ROWS_*12*4;
  float*          spb  = (float*)(w + o);          o += (size_t)B_*N_*N_*4;   // 32 MiB (L2-resident)
  unsigned*       adj  = (unsigned*)(w + o);       o += (size_t)B_*N_*N_/8;   // 1 MiB

  // pack / precompute
  cvt_bf16_kernel<<<(ROWS_*D_ + 255)/256, 256, 0, stream>>>(nodes, nbf, ROWS_*D_);
  geo_kernel<<<(ROWS_ + 255)/256, 256, 0, stream>>>(boxes, ngeo);
  pack_qkvT_kernel<<<(768*256 + 255)/256, 256, 0, stream>>>(Wq, Wk, Wv, bq, bk, bv, WqkvT, bqkv);
  packT_kernel<<<(256*256 + 255)/256, 256, 0, stream>>>(Wo, WoT, 256, 256, 1.f);
  packT_kernel<<<(256*1024 + 255)/256, 256, 0, stream>>>(W1, W1T, 256, 1024, 1.f);
  packT_kernel<<<(1024*256 + 255)/256, 256, 0, stream>>>(W2, W2T, 1024, 256, 1.f);

  // sparse bias / adjacency
  const int n4 = (int)(((size_t)B_*N_*N_*4 + (size_t)B_*N_*N_/8) / 16);
  zero_kernel<<<2048, 256, 0, stream>>>((uint4*)spb, n4);
  scatter_kernel<<<(B_*E_ + 255)/256, 256, 0, stream>>>(eidx, eattr, edge_w, edge_b, spb, adj);

  // main pipeline
  qkv_kernel<<<dim3(ROWS_/64, 12), 128, 0, stream>>>(nbf, WqkvT, bqkv, qbf, kbf, vbf);
  attn_kernel<<<dim3(N_/64, B_), 128, 0, stream>>>(qbf, kbf, vbf, ngeo, spb, adj,
                                                   nmask, geo_w, geo_b, abf);
  proj_ln1_kernel<<<ROWS_/64, 128, 0, stream>>>(abf, WoT, bo, nodes, ln1g, ln1b, xf32, xbf);
  ffn_kernel<<<ROWS_/64, 128, 0, stream>>>(xbf, xf32, W1T, b1f, W2T, b2f,
                                           ln2g, ln2b, nmask, out);
}
